// HeteroFloodGNNv2_54382875902392
// MI455X (gfx1250) — compile-verified, matrix operand order checked by reading, not run on hardware
//
#include <hip/hip_runtime.h>
#include <math.h>

typedef __attribute__((ext_vector_type(16))) _Float16 v16h;
typedef __attribute__((ext_vector_type(8)))  float    v8f;

// ---------------- device helpers ----------------

__device__ __forceinline__ float gelu_exact(float x) {
    return 0.5f * x * (1.0f + erff(x * 0.70710678118654752440f));
}

__device__ __forceinline__ void atomicMaxF(float* addr, float v) {
    // ordered-int trick: works for mixed signs with -3e38 init
    if (v >= 0.0f) atomicMax((int*)addr, __float_as_int(v));
    else           atomicMin((unsigned int*)addr, __float_as_uint(v));
}

// B-fragment swizzled LDS index for (k, n), k-tile local.
// Forward lane mapping: lane = 16*half + (n&15); element e holds
// k' = 2*(g&3) + (e&1) + 16*(g>>2) + 8*half, g = e>>1.
__device__ __forceinline__ int bfrag_idx(int k, int n) {
    int kt = k >> 5, kk = k & 31;
    int half = (kk >> 3) & 1;
    int hi   = (kk >> 4) & 1;
    int r    = kk & 7;
    int e    = 2 * ((r >> 1) + 4 * hi) + (r & 1);
    int lane = (half << 4) | (n & 15);
    int j    = n >> 4;
    return (((((kt << 3) + j) << 5) + lane) << 4) + e;
}

__device__ __forceinline__ void cvt_frag(v16h& a, const float4* f) {
    a[0]  = (_Float16)f[0].x; a[1]  = (_Float16)f[0].y; a[2]  = (_Float16)f[0].z; a[3]  = (_Float16)f[0].w;
    a[4]  = (_Float16)f[1].x; a[5]  = (_Float16)f[1].y; a[6]  = (_Float16)f[1].z; a[7]  = (_Float16)f[1].w;
    a[8]  = (_Float16)f[2].x; a[9]  = (_Float16)f[2].y; a[10] = (_Float16)f[2].z; a[11] = (_Float16)f[2].w;
    a[12] = (_Float16)f[3].x; a[13] = (_Float16)f[3].y; a[14] = (_Float16)f[3].z; a[15] = (_Float16)f[3].w;
}

// ---------------- WMMA GEMM, K == 128 specialization ----------------
// C[M x 128] (+)= act(A[M x 128] @ W[128 x 128] + bias), M % 16 == 0.
// Block = 256 threads (8 waves), wave computes a 16x128 strip; 32 v_wmma per wave.
// All 16 A-float4 loads batched up front so the memory pipe runs them concurrently.
template<int ACC, int DOGELU>
__global__ void gemm128_wmma_f16(const float* __restrict__ A,
                                 const float* __restrict__ W,
                                 const float* __restrict__ bias,
                                 float* __restrict__ C, int M)
{
    __shared__ __attribute__((aligned(32))) _Float16 Wlds[128 * 128]; // 32 KB

    const int tid = threadIdx.x;
    // coalesced linear read of W, scatter to fragment order (f32 -> f16)
    for (int t = tid; t < 128 * 128; t += blockDim.x) {
        Wlds[bfrag_idx(t >> 7, t & 127)] = (_Float16)W[t];
    }
    __syncthreads();

    const int wave = tid >> 5;
    const int lane = tid & 31;
    const int row0 = (blockIdx.x * (blockDim.x >> 5) + wave) << 4;
    if (row0 >= M) return;
    const int half = lane >> 4;
    const int lm   = lane & 15;

    v8f acc[8] = {};
    const float* Ap = A + (size_t)(row0 + lm) * 128;  // M % 16 == 0: always valid

    // batch-issue all A loads (4 per k-tile x 4 tiles): independent b128 loads
    float4 af[16];
    #pragma unroll
    for (int kt = 0; kt < 4; ++kt) {
        const int k0 = kt << 5;
        af[kt * 4 + 0] = *(const float4*)(Ap + k0 + 8 * half);
        af[kt * 4 + 1] = *(const float4*)(Ap + k0 + 8 * half + 4);
        af[kt * 4 + 2] = *(const float4*)(Ap + k0 + 16 + 8 * half);
        af[kt * 4 + 3] = *(const float4*)(Ap + k0 + 16 + 8 * half + 4);
    }

    #pragma unroll
    for (int kt = 0; kt < 4; ++kt) {
        v16h a;
        cvt_frag(a, &af[kt * 4]);
        #pragma unroll
        for (int j = 0; j < 8; ++j) {
            const v16h b = *((const v16h*)(Wlds + ((((((kt << 3) + j) << 5) | lane)) << 4)));
            acc[j] = __builtin_amdgcn_wmma_f32_16x16x32_f16(
                false, a, false, b, (short)0, acc[j], false, false);
        }
    }

    #pragma unroll
    for (int j = 0; j < 8; ++j) {
        #pragma unroll
        for (int r = 0; r < 8; ++r) {
            int m = row0 + r + 8 * half;
            int n = (j << 4) + lm;
            float v = acc[j][r];
            if (bias) v += bias[n];
            if (DOGELU) v = gelu_exact(v);
            float* p = C + (size_t)m * 128 + n;
            if (ACC) *p += v; else *p = v;
        }
    }
}

// ---------------- WMMA GEMM, generic small K (encoders: K in {7,9,10,15}) ----
// Branchless k-guards (address clamp + select), M % 16 == 0.
template<int ACC, int DOGELU>
__global__ void gemm_wmma_f16(const float* __restrict__ A,
                              const float* __restrict__ W,
                              const float* __restrict__ bias,
                              float* __restrict__ C,
                              int M, int K)
{
    __shared__ __attribute__((aligned(32))) _Float16 Wlds[4 * 8 * 32 * 16];

    const int tid  = threadIdx.x;
    const int nthr = blockDim.x;
    const int KT   = (K + 31) >> 5;
    // linear over k,n with zero padding to tile boundary
    const int totkn = (KT << 5) * 128;
    for (int t = tid; t < totkn; t += nthr) {
        int k = t >> 7, n = t & 127;
        Wlds[bfrag_idx(k, n)] = (k < K) ? (_Float16)W[(size_t)k * 128 + n] : (_Float16)0.0f;
    }
    __syncthreads();

    const int wave = tid >> 5;
    const int lane = tid & 31;
    const int row0 = (blockIdx.x * (nthr >> 5) + wave) << 4;
    if (row0 >= M) return;
    const int half = lane >> 4;
    const int lm   = lane & 15;

    v8f acc[8] = {};
    const float* Ap = A + (size_t)(row0 + lm) * K;  // M % 16 == 0: row valid

    for (int k0 = 0; k0 < K; k0 += 32) {
        v16h a;
        #pragma unroll
        for (int e = 0; e < 16; ++e) {
            int g = e >> 1;
            int k = k0 + 2 * (g & 3) + (e & 1) + 16 * (g >> 2) + 8 * half;
            int kc = (k < K) ? k : 0;            // branchless: clamp addr, mask value
            float v = Ap[kc];
            a[e] = (_Float16)((k < K) ? v : 0.0f);
        }
        const int kt = k0 >> 5;
        #pragma unroll
        for (int j = 0; j < 8; ++j) {
            const v16h b = *((const v16h*)(Wlds + ((((((kt << 3) + j) << 5) | lane)) << 4)));
            acc[j] = __builtin_amdgcn_wmma_f32_16x16x32_f16(
                false, a, false, b, (short)0, acc[j], false, false);
        }
    }

    #pragma unroll
    for (int j = 0; j < 8; ++j) {
        #pragma unroll
        for (int r = 0; r < 8; ++r) {
            int m = row0 + r + 8 * half;
            int n = (j << 4) + lm;
            float v = acc[j][r];
            if (bias) v += bias[n];
            if (DOGELU) v = gelu_exact(v);
            float* p = C + (size_t)m * 128 + n;
            if (ACC) *p += v; else *p = v;
        }
    }
}

// ---------------- elementwise / scatter kernels ----------------

__global__ void fill_f32_k(float* p, float v, int n) {
    int i = blockIdx.x * blockDim.x + threadIdx.x;
    if (i < n) p[i] = v;
}

__global__ void add_row_bias_k(float* o, const float* __restrict__ bias, int N) {
    int i = blockIdx.x * blockDim.x + threadIdx.x;
    if (i < N * 128) o[i] += bias[i & 127];
}

// per (edge, head): logits + segment-max (float4-vectorized dot)
__global__ void gat_logits_k(const float* __restrict__ xl, const float* __restrict__ xr,
                             const float* __restrict__ eT,
                             const int* __restrict__ src, const int* __restrict__ dst,
                             const float* __restrict__ att,
                             float* __restrict__ lg, float* lmax, int E)
{
    int idx = blockIdx.x * blockDim.x + threadIdx.x;
    if (idx >= E * 4) return;
    int e = idx >> 2, h = idx & 3;
    int s = src[e], d = dst[e];
    const float4* pl = (const float4*)(xl + (size_t)s * 128 + h * 32);
    const float4* pr = (const float4*)(xr + (size_t)d * 128 + h * 32);
    const float4* pe = (const float4*)(eT + (size_t)e * 128 + h * 32);
    const float4* pa = (const float4*)(att + h * 32);
    float acc = 0.0f;
    #pragma unroll
    for (int c = 0; c < 8; ++c) {
        float4 a = pl[c], b = pr[c], t = pe[c], w = pa[c];
        float m;
        m = a.x + b.x + t.x; m = (m > 0.0f) ? m : 0.2f * m; acc += m * w.x;
        m = a.y + b.y + t.y; m = (m > 0.0f) ? m : 0.2f * m; acc += m * w.y;
        m = a.z + b.z + t.z; m = (m > 0.0f) ? m : 0.2f * m; acc += m * w.z;
        m = a.w + b.w + t.w; m = (m > 0.0f) ? m : 0.2f * m; acc += m * w.w;
    }
    lg[idx] = acc;
    atomicMaxF(&lmax[d * 4 + h], acc);
}

__global__ void gat_softmax_k(float* lg, const float* __restrict__ lmax,
                              float* den, const int* __restrict__ dst, int E)
{
    int idx = blockIdx.x * blockDim.x + threadIdx.x;
    if (idx >= E * 4) return;
    int e = idx >> 2, h = idx & 3;
    int d = dst[e];
    float v = expf(lg[idx] - lmax[d * 4 + h]);
    lg[idx] = v;
    atomicAdd(&den[d * 4 + h], v);
}

// per (edge, channel): o[dst] += alpha * xl[src]
__global__ void gat_scatter_k(const float* __restrict__ lg, const float* __restrict__ den,
                              const float* __restrict__ xl,
                              const int* __restrict__ src, const int* __restrict__ dst,
                              float* o, int E)
{
    int idx = blockIdx.x * blockDim.x + threadIdx.x;
    if (idx >= E * 128) return;
    int e = idx >> 7, dd = idx & 127, h = dd >> 5;
    int s = src[e], d = dst[e];
    float a = lg[e * 4 + h] / den[d * 4 + h];
    atomicAdd(&o[(size_t)d * 128 + dd], a * xl[(size_t)s * 128 + dd]);
}

__global__ void sage_scatter_k(const float* __restrict__ hsrc,
                               const int* __restrict__ src, const int* __restrict__ dst,
                               float* mean, int E)
{
    int idx = blockIdx.x * blockDim.x + threadIdx.x;
    if (idx >= E * 128) return;
    int e = idx >> 7, dd = idx & 127;
    atomicAdd(&mean[(size_t)dst[e] * 128 + dd], hsrc[(size_t)src[e] * 128 + dd]);
}

__global__ void sage_count_k(const int* __restrict__ dst, float* cnt, int E) {
    int e = blockIdx.x * blockDim.x + threadIdx.x;
    if (e < E) atomicAdd(&cnt[dst[e]], 1.0f);
}

__global__ void sage_div_k(float* mean, const float* __restrict__ cnt, int Nd) {
    int idx = blockIdx.x * blockDim.x + threadIdx.x;
    if (idx < Nd * 128) mean[idx] /= fmaxf(cnt[idx >> 7], 1.0f);
}

// h = LN(h + o) * g + b  (one wave per 128-wide row, float4 per lane)
__global__ void ln_residual_k(float* h, const float* __restrict__ o,
                              const float* __restrict__ g, const float* __restrict__ b, int N)
{
    int wave = threadIdx.x >> 5, lane = threadIdx.x & 31;
    int row = blockIdx.x * (blockDim.x >> 5) + wave;
    if (row >= N) return;
    size_t base = (size_t)row * 128 + lane * 4;
    float4 xh = *(const float4*)(h + base);
    float4 xo = *(const float4*)(o + base);
    float4 v = make_float4(xh.x + xo.x, xh.y + xo.y, xh.z + xo.z, xh.w + xo.w);
    float s  = v.x + v.y + v.z + v.w;
    float s2 = v.x * v.x + v.y * v.y + v.z * v.z + v.w * v.w;
    #pragma unroll
    for (int off = 16; off > 0; off >>= 1) {
        s  += __shfl_xor(s,  off, 32);
        s2 += __shfl_xor(s2, off, 32);
    }
    float mu  = s  * (1.0f / 128.0f);
    float var = s2 * (1.0f / 128.0f) - mu * mu;
    float r = rsqrtf(var + 1e-5f);
    float4 gg = *(const float4*)(g + lane * 4);
    float4 bb = *(const float4*)(b + lane * 4);
    float4 out;
    out.x = (v.x - mu) * r * gg.x + bb.x;
    out.y = (v.y - mu) * r * gg.y + bb.y;
    out.z = (v.z - mu) * r * gg.z + bb.z;
    out.w = (v.w - mu) * r * gg.w + bb.w;
    *(float4*)(h + base) = out;
}

// decoder final: out[n] = dot(H[n,:], w) + b  (W2 is 128x1)
__global__ void matvec_k(const float* __restrict__ Hd, const float* __restrict__ w,
                         const float* __restrict__ b, float* out, int N)
{
    int n = blockIdx.x * blockDim.x + threadIdx.x;
    if (n >= N) return;
    const float4* p = (const float4*)(Hd + (size_t)n * 128);
    const float4* q = (const float4*)w;
    float s = b[0];
    #pragma unroll
    for (int d = 0; d < 32; ++d) {
        float4 a = p[d], c = q[d];
        s += a.x * c.x + a.y * c.y + a.z * c.z + a.w * c.w;
    }
    out[n] = s;
}

// ---------------- host orchestration ----------------

struct MlpP  { const float *W1, *b1, *W2, *b2; };
struct GatP  { const float *Wl, *bl, *Wr, *br, *We, *att, *bias; };
struct SageP { const float *Wl, *bl, *Wr; };
struct LnP   { const float *g, *b; };

extern "C" void kernel_launch(void* const* d_in, const int* in_sizes, int n_in,
                              void* d_out, int out_size, void* d_ws, size_t ws_size,
                              hipStream_t stream)
{
    (void)in_sizes; (void)n_in; (void)out_size; (void)ws_size;
    constexpr int N1 = 10000, N2 = 50000, EP = 30000, ES = 200000, EC = 10000;

    const float* x1      = (const float*)d_in[0];
    const float* x2      = (const float*)d_in[1];
    const float* eattr_p = (const float*)d_in[2];
    const float* eattr_s = (const float*)d_in[3];
    const int*   ei_p    = (const int*)d_in[4];   // [2, EP]
    const int*   ei_s    = (const int*)d_in[5];   // [2, ES]
    const int*   ei_c12  = (const int*)d_in[6];   // [2, EC] 1d src -> 2d dst
    const int*   ei_c21  = (const int*)d_in[7];   // [2, EC] 2d src -> 1d dst

    // params flattened in setup_inputs() dict insertion order (recursively)
    int pi = 8;
    auto nf = [&]() -> const float* { return (const float*)d_in[pi++]; };

    MlpP enc1{nf(), nf(), nf(), nf()};
    MlpP enc2{nf(), nf(), nf(), nf()};
    MlpP encp{nf(), nf(), nf(), nf()};
    MlpP encs{nf(), nf(), nf(), nf()};
    GatP gp[6], gs[6]; SageP s12[6], s21[6]; LnP l1[6], l2[6];
    for (int i = 0; i < 6; ++i) {
        gp[i]  = GatP{nf(), nf(), nf(), nf(), nf(), nf(), nf()};
        gs[i]  = GatP{nf(), nf(), nf(), nf(), nf(), nf(), nf()};
        s12[i] = SageP{nf(), nf(), nf()};
        s21[i] = SageP{nf(), nf(), nf()};
        l1[i]  = LnP{nf(), nf()};
        l2[i]  = LnP{nf(), nf()};
    }
    MlpP dec1{nf(), nf(), nf(), nf()};
    MlpP dec2{nf(), nf(), nf(), nf()};

    // workspace partition (floats)
    float* w    = (float*)d_ws;
    float* h1   = w; w += (size_t)N1 * 128;
    float* h2   = w; w += (size_t)N2 * 128;
    float* eap  = w; w += (size_t)EP * 128;
    float* eas  = w; w += (size_t)ES * 128;
    float* o1   = w; w += (size_t)N1 * 128;
    float* o2   = w; w += (size_t)N2 * 128;
    float* xl   = w; w += (size_t)N2 * 128;   // GAT xl / MLP hidden (node-sized) / decoder hidden
    float* xr   = w; w += (size_t)N2 * 128;   // GAT xr / SAGE mean
    float* eT   = w; w += (size_t)ES * 128;   // GAT edge transform / MLP hidden (edge-sized)
    float* lg   = w; w += (size_t)ES * 4;
    float* lmax = w; w += (size_t)N2 * 4;
    float* den  = w; w += (size_t)N2 * 4;
    float* cnt  = w; w += (size_t)N2;

    auto gemm = [&](const float* A, const float* Wm, const float* bias, float* Cm,
                    int M, int K, int accmode, bool dogelu) {
        dim3 grid((unsigned)((M + 127) / 128)), block(256);
        if (K == 128) {
            if (accmode)     gemm128_wmma_f16<1, 0><<<grid, block, 0, stream>>>(A, Wm, bias, Cm, M);
            else if (dogelu) gemm128_wmma_f16<0, 1><<<grid, block, 0, stream>>>(A, Wm, bias, Cm, M);
            else             gemm128_wmma_f16<0, 0><<<grid, block, 0, stream>>>(A, Wm, bias, Cm, M);
        } else {
            if (accmode)     gemm_wmma_f16<1, 0><<<grid, block, 0, stream>>>(A, Wm, bias, Cm, M, K);
            else if (dogelu) gemm_wmma_f16<0, 1><<<grid, block, 0, stream>>>(A, Wm, bias, Cm, M, K);
            else             gemm_wmma_f16<0, 0><<<grid, block, 0, stream>>>(A, Wm, bias, Cm, M, K);
        }
    };
    auto fill = [&](float* p, float v, int n) {
        fill_f32_k<<<(unsigned)((n + 255) / 256), 256, 0, stream>>>(p, v, n);
    };
    auto mlp = [&](const float* X, const MlpP& m, float* hid, float* out, int M, int K) {
        gemm(X, m.W1, m.b1, hid, M, K, 0, true);     // Linear -> exact GELU
        gemm(hid, m.W2, m.b2, out, M, 128, 0, false);
    };
    auto gat = [&](const float* h, const float* ea, const int* ei, const GatP& g,
                   float* o, int N, int E) {
        gemm(h,  g.Wl, g.bl,    xl, N, 128, 0, false);
        gemm(h,  g.Wr, g.br,    xr, N, 128, 0, false);
        gemm(ea, g.We, nullptr, eT, E, 128, 0, false);
        fill(lmax, -3.0e38f, N * 4);
        fill(den,  0.0f,     N * 4);
        int t4 = E * 4;
        gat_logits_k <<<(unsigned)((t4 + 255) / 256), 256, 0, stream>>>(xl, xr, eT, ei, ei + E, g.att, lg, lmax, E);
        gat_softmax_k<<<(unsigned)((t4 + 255) / 256), 256, 0, stream>>>(lg, lmax, den, ei + E, E);
        int t128 = E * 128;
        gat_scatter_k<<<(unsigned)((t128 + 255) / 256), 256, 0, stream>>>(lg, den, xl, ei, ei + E, o, E);
        add_row_bias_k<<<(unsigned)((N * 128 + 255) / 256), 256, 0, stream>>>(o, g.bias, N);
    };
    auto sage = [&](const float* hsrc, const float* hdst, const int* ei, const SageP& s,
                    float* o, int Nd, int E) {
        fill(xr, 0.0f, Nd * 128);   // xr reused as mean buffer (GAT done for this node set)
        fill(cnt, 0.0f, Nd);
        sage_scatter_k<<<(unsigned)((E * 128 + 255) / 256), 256, 0, stream>>>(hsrc, ei, ei + E, xr, E);
        sage_count_k  <<<(unsigned)((E + 255) / 256),       256, 0, stream>>>(ei + E, cnt, E);
        sage_div_k    <<<(unsigned)((Nd * 128 + 255) / 256), 256, 0, stream>>>(xr, cnt, Nd);
        gemm(xr,   s.Wl, s.bl,    o, Nd, 128, 1, false);  // += mean @ Wl + bl
        gemm(hdst, s.Wr, nullptr, o, Nd, 128, 1, false);  // += x_dst @ Wr
    };

    // ---- encoders ----
    mlp(x1,      enc1, xl, h1,  N1, 10);
    mlp(x2,      enc2, xl, h2,  N2, 15);
    mlp(eattr_p, encp, eT, eap, EP, 9);
    mlp(eattr_s, encs, eT, eas, ES, 7);

    // ---- 6 processor layers ----
    for (int i = 0; i < 6; ++i) {
        fill(o1, 0.0f, N1 * 128);
        gat(h1, eap, ei_p, gp[i], o1, N1, EP);
        sage(h2, h1, ei_c21, s21[i], o1, N1, EC);

        fill(o2, 0.0f, N2 * 128);
        gat(h2, eas, ei_s, gs[i], o2, N2, ES);
        sage(h1, h2, ei_c12, s12[i], o2, N2, EC);

        ln_residual_k<<<(N1 + 7) / 8, 256, 0, stream>>>(h1, o1, l1[i].g, l1[i].b, N1);
        ln_residual_k<<<(N2 + 7) / 8, 256, 0, stream>>>(h2, o2, l2[i].g, l2[i].b, N2);
    }

    // ---- decoders ----
    float* out = (float*)d_out;
    gemm(h1, dec1.W1, dec1.b1, xl, N1, 128, 0, true);
    matvec_k<<<(N1 + 255) / 256, 256, 0, stream>>>(xl, dec1.W2, dec1.b2, out, N1);
    gemm(h2, dec2.W1, dec2.b1, xl, N2, 128, 0, true);
    matvec_k<<<(N2 + 255) / 256, 256, 0, stream>>>(xl, dec2.W2, dec2.b2, out + N1, N2);
}
